// HierarchicalFormationTransformer_29489245454854
// MI455X (gfx1250) — matrix operand order repaired and checked by reading, not compiled, not run on hardware
//
#include <hip/hip_runtime.h>
#include <hip/hip_bf16.h>
#include <math.h>

#define LAYERS 4
#define BB 8
#define SS 1024
#define EE 512
#define HH 8
#define DD 64
#define FFN 2048
#define MROWS (BB*SS) /* 8192 */

typedef __bf16 bf16;
typedef __attribute__((ext_vector_type(16))) __bf16 v16bf;
typedef __attribute__((ext_vector_type(8)))  float  v8f;

union AF { v16bf v; uint4 q[2]; };

__device__ __forceinline__ v8f wmma_bf16(v16bf a, v16bf b, v8f c) {
  // (neg_a, A, neg_b, B, c_mod, C, reuse_a, reuse_b)
  return __builtin_amdgcn_wmma_f32_16x16x32_bf16(false, a, false, b, (short)0, c, false, false);
}

// ---------------------------------------------------------------- TDM (Tensor Data Mover)
#if defined(__has_builtin)
#  if __has_builtin(__builtin_amdgcn_tensor_load_to_lds)
#    define HAVE_TDM 1
#  endif
#endif

#ifdef HAVE_TDM
typedef unsigned int u32x4 __attribute__((ext_vector_type(4)));
typedef int          i32x4 __attribute__((ext_vector_type(4)));
typedef int          i32x8 __attribute__((ext_vector_type(8)));

// 2D tile DMA: global (row-major, elem=2B, row stride = stride0 elems) -> LDS.
// If pad!=0, insert 4 DWORDs (8 bf16) of LDS padding after every 16 DWORDs
// (64B = one 32-elem row) -> LDS pitch 40 elems, matching At[128][40].
__device__ __forceinline__ void tdm_load_tile_2d(
    unsigned int lds_off, const void* gptr,
    unsigned int td0, unsigned int td1,
    unsigned int tile0, unsigned int tile1,
    unsigned long long stride0, int pad) {
  unsigned long long ga = (unsigned long long)(uintptr_t)gptr;
  u32x4 g0;
  g0.x = 1u;                                                 // count=1 (valid), user mode
  g0.y = lds_off;                                            // lds_addr (bytes)
  g0.z = (unsigned int)ga;                                   // global_addr[31:0]
  g0.w = (unsigned int)((ga >> 32) & 0x1FFFFFFu) | (2u << 30); // addr[56:32] | type=2
  i32x8 g1;
  unsigned int flags = (1u << 16);                           // data_size = 2 bytes
  if (pad) flags |= (1u << 20) | (3u << 22) | (3u << 25);    // pad_en, interval=16dw, amount=4dw
  g1[0] = (int)flags;                                        // wg_mask=0 (not in cluster)
  g1[1] = (int)((td0 & 0xFFFFu) << 16);                      // tensor_dim0[15:0] @ 63:48
  g1[2] = (int)((td0 >> 16) | ((td1 & 0xFFFFu) << 16));      // td0[31:16] | td1[15:0]
  g1[3] = (int)((td1 >> 16) | (tile0 << 16));                // td1[31:16] | tile_dim0
  g1[4] = (int)(tile1 & 0xFFFFu);                            // tile_dim1 | tile_dim2=0
  g1[5] = (int)(unsigned int)stride0;                        // dim0_stride[31:0] (elems)
  g1[6] = (int)((unsigned int)(stride0 >> 32) & 0xFFFFu);    // dim0_stride[47:32] | d1s[15:0]=0
  g1[7] = 0;
  i32x4 z4 = {0, 0, 0, 0};
  i32x8 z8 = {0, 0, 0, 0, 0, 0, 0, 0};
  __builtin_amdgcn_tensor_load_to_lds(g0, g1, z4, z4, z8, 0);
}
#endif

// ---------------------------------------------------------------- cvt f32->bf16
__global__ __launch_bounds__(256)
void cvt_kernel(const float* __restrict__ s, bf16* __restrict__ d, size_t n) {
  size_t i = (size_t)blockIdx.x * blockDim.x + threadIdx.x;
  size_t stride = (size_t)gridDim.x * blockDim.x;
  for (; i < n; i += stride) d[i] = (bf16)s[i];
}

// ---------------------------------------------------------------- layernorm (row of 512)
__global__ __launch_bounds__(256)
void layernorm_kernel(const float* __restrict__ x, const float* __restrict__ g,
                      const float* __restrict__ bt, bf16* __restrict__ out) {
  __shared__ float red[8];
  __shared__ float stat[2];
  const int row = blockIdx.x, tid = threadIdx.x;
  const int lane = tid & 31, w = tid >> 5;
  const float* xr = x + (size_t)row * EE;
  float a0 = xr[tid], a1 = xr[tid + 256];
  float s = a0 + a1;
  for (int m = 1; m < 32; m <<= 1) s += __shfl_xor(s, m, 32);
  if (lane == 0) red[w] = s;
  __syncthreads();
  if (tid == 0) {
    float t = 0.f;
    for (int i = 0; i < 8; i++) t += red[i];
    stat[0] = t * (1.0f / EE);
  }
  __syncthreads();
  const float mu = stat[0];
  float d0 = a0 - mu, d1 = a1 - mu;
  float s2 = d0 * d0 + d1 * d1;
  for (int m = 1; m < 32; m <<= 1) s2 += __shfl_xor(s2, m, 32);
  __syncthreads();
  if (lane == 0) red[w] = s2;
  __syncthreads();
  if (tid == 0) {
    float t = 0.f;
    for (int i = 0; i < 8; i++) t += red[i];
    stat[1] = rsqrtf(t * (1.0f / EE) + 1e-5f);
  }
  __syncthreads();
  const float rs = stat[1];
  out[(size_t)row * EE + tid]       = (bf16)(d0 * rs * g[tid] + bt[tid]);
  out[(size_t)row * EE + tid + 256] = (bf16)(d1 * rs * g[tid + 256] + bt[tid + 256]);
}

// ---------------------------------------------------------------- bf16 GEMM, f32 accum
// C[M,N] = epilogue(A[M,K] @ B[K,N] + bias [+ residual] [GELU]) -> outF (f32) / outB (bf16)
__global__ __launch_bounds__(256)
void gemm_bf16_kernel(const bf16* __restrict__ A, const bf16* __restrict__ Bw,
                      const float* __restrict__ bias, const float* __restrict__ resid,
                      float* __restrict__ outF, bf16* __restrict__ outB,
                      int M, int N, int K, int gelu) {
  __shared__ __align__(16) bf16 At[128][40];   // row-major A tile, TDM-padded pitch
  __shared__ __align__(16) bf16 Bt[128][40];   // B tile stored transposed [n][k]
  const int tid = threadIdx.x;
  const int lane = tid & 31, w = tid >> 5;
  const int hl = lane >> 4, l15 = lane & 15;
  const int wm = (w >> 1) * 32, wn = (w & 1) * 64;
  const int m0 = blockIdx.y * 128, n0 = blockIdx.x * 128;

  v8f acc[2][4];
#pragma unroll
  for (int i = 0; i < 2; i++)
#pragma unroll
    for (int j = 0; j < 4; j++) { v8f z = {}; acc[i][j] = z; }

  for (int k0 = 0; k0 < K; k0 += 32) {
    // ---- stage A tile: TDM DMA (wave 0 issues; padding gives 40-elem pitch)
#ifdef HAVE_TDM
    if (tid < 32) {
      tdm_load_tile_2d((unsigned int)(uintptr_t)&At[0][0],
                       A + (size_t)m0 * K + k0,
                       (unsigned int)K, (unsigned int)M,
                       32u, 128u, (unsigned long long)K, 1);
    }
#else
#pragma unroll
    for (int i = 0; i < 2; i++) {
      int c = tid + i * 256;
      int r = c >> 2, kc = (c & 3) * 8;
      *(uint4*)&At[r][kc] = *(const uint4*)(A + (size_t)(m0 + r) * K + k0 + kc);
    }
#endif
    // ---- stage B transposed: each thread walks 16 k's of one column
    {
      int n = tid & 127;
      int kh = (tid >> 7) * 16;
      if (k0 + 32 < K)  // prefetch next B tile rows (-> global_prefetch_b8)
        __builtin_prefetch(Bw + (size_t)(k0 + 32 + kh) * N + n0 + n, 0, 1);
      bf16 tmp[16];
#pragma unroll
      for (int j = 0; j < 16; j++)
        tmp[j] = Bw[(size_t)(k0 + kh + j) * N + n0 + n];
      *(uint4*)&Bt[n][kh]     = *(uint4*)&tmp[0];
      *(uint4*)&Bt[n][kh + 8] = *(uint4*)&tmp[8];
    }
#ifdef HAVE_TDM
    if (tid < 32) __builtin_amdgcn_s_wait_tensorcnt(0);
#endif
    __syncthreads();

    AF af[2], bf[4];
#pragma unroll
    for (int mf = 0; mf < 2; mf++) {
      const bf16* pr = &At[wm + mf * 16 + l15][hl * 8];
      af[mf].q[0] = *(const uint4*)pr;          // K = hl*8 .. +7
      af[mf].q[1] = *(const uint4*)(pr + 16);   // K = 16+hl*8 .. +7
    }
#pragma unroll
    for (int nf = 0; nf < 4; nf++) {
      const bf16* pc = &Bt[wn + nf * 16 + l15][hl * 8];
      bf[nf].q[0] = *(const uint4*)pc;          // K = hl*8 .. +7
      bf[nf].q[1] = *(const uint4*)(pc + 16);   // K = 16+hl*8 .. +7
    }
#pragma unroll
    for (int mf = 0; mf < 2; mf++)
#pragma unroll
      for (int nf = 0; nf < 4; nf++)
        acc[mf][nf] = wmma_bf16(af[mf].v, bf[nf].v, acc[mf][nf]);
    __syncthreads();
  }

  // epilogue: C layout -> lane<16: (M=r,N=l15); lane>=16: (M=r+8,N=l15)
#pragma unroll
  for (int mf = 0; mf < 2; mf++)
#pragma unroll
    for (int nf = 0; nf < 4; nf++) {
      const int ncol = n0 + wn + nf * 16 + l15;
      const float bv = bias[ncol];
#pragma unroll
      for (int r = 0; r < 8; r++) {
        const int mrow = m0 + wm + mf * 16 + r + 8 * hl;
        const size_t idx = (size_t)mrow * N + ncol;
        float v = acc[mf][nf][r] + bv;
        if (resid) v += resid[idx];
        if (gelu)  v = 0.5f * v * (1.0f + erff(v * 0.70710678f));
        if (outF)  outF[idx] = v;
        if (outB)  outB[idx] = (bf16)v;
      }
    }
}

// ---------------------------------------------------------------- flash attention (bf16 WMMA)
__global__ __launch_bounds__(128)
void attention_kernel(const bf16* __restrict__ Qb, const bf16* __restrict__ Kb,
                      const bf16* __restrict__ Vb, const float* __restrict__ ebias,
                      const unsigned char* __restrict__ kpm, bf16* __restrict__ outB) {
  __shared__ __align__(16) bf16 Kt[32][64];      // [key][d]
  __shared__ __align__(16) bf16 Vt[64][32];      // transposed [d][key]
  __shared__ __align__(16) bf16 Pb[4][16][40];   // per-wave P staging
  const int tid = threadIdx.x;
  const int lane = tid & 31, w = tid >> 5;
  const int hl = lane >> 4, l15 = lane & 15;
  const int blk = blockIdx.x;
  const int qb = blk & 15;          // S/64 = 16
  const int h  = (blk >> 4) & 7;
  const int b  = blk >> 7;
  const int q0 = qb * 64 + w * 16;

  // Q fragments: 16x64, split in two 16x32 A-frags
  AF qf[2];
  {
    const bf16* qp = Qb + ((size_t)(b * SS + q0 + l15)) * EE + h * DD;
#pragma unroll
    for (int f = 0; f < 2; f++) {
      qf[f].q[0] = *(const uint4*)(qp + f * 32 + hl * 8);
      qf[f].q[1] = *(const uint4*)(qp + f * 32 + 16 + hl * 8);
    }
  }
  float ml[8], ll[8];
  v8f O[4];
#pragma unroll
  for (int r = 0; r < 8; r++) { ml[r] = -1e30f; ll[r] = 0.f; }
#pragma unroll
  for (int dt = 0; dt < 4; dt++) { v8f z = {}; O[dt] = z; }

  for (int s0 = 0; s0 < SS; s0 += 32) {
    // ---- stage K tile: TDM DMA (dense 128B rows, no padding)
#ifdef HAVE_TDM
    if (tid < 32) {
      tdm_load_tile_2d((unsigned int)(uintptr_t)&Kt[0][0],
                       Kb + ((size_t)(b * SS + s0)) * EE + h * DD,
                       (unsigned int)EE, (unsigned int)(BB * SS),
                       64u, 32u, (unsigned long long)EE, 0);
    }
#else
#pragma unroll
    for (int i = 0; i < 2; i++) {
      int c = tid + i * 128;
      int key = c >> 3, kc = (c & 7) * 8;
      *(uint4*)&Kt[key][kc] =
          *(const uint4*)(Kb + ((size_t)(b * SS + s0 + key)) * EE + h * DD + kc);
    }
#endif
    // ---- stage V transposed
#pragma unroll
    for (int i = 0; i < 16; i++) {
      int idx = tid + i * 128;
      int d = idx >> 5, key = idx & 31;
      Vt[d][key] = Vb[((size_t)(b * SS + s0 + key)) * EE + h * DD + d];
    }
#ifdef HAVE_TDM
    if (tid < 32) __builtin_amdgcn_s_wait_tensorcnt(0);
#endif
    __syncthreads();

    // scores: two 16x16 tiles over this 32-key block
    v8f sc[2];
#pragma unroll
    for (int tk = 0; tk < 2; tk++) {
      v8f z = {}; sc[tk] = z;
#pragma unroll
      for (int f = 0; f < 2; f++) {
        AF kf;
        const bf16* kp = &Kt[tk * 16 + l15][f * 32 + hl * 8];
        kf.q[0] = *(const uint4*)kp;
        kf.q[1] = *(const uint4*)(kp + 16);
        sc[tk] = wmma_bf16(qf[f].v, kf.v, sc[tk]);
      }
    }
    // scale, pad-mask, edge bias (reference order)
#pragma unroll
    for (int tk = 0; tk < 2; tk++) {
      const int key = s0 + tk * 16 + l15;
      const bool msk = kpm[b * SS + key] != 0;
#pragma unroll
      for (int r = 0; r < 8; r++) {
        const int q = q0 + r + 8 * hl;
        float v = sc[tk][r] * 0.125f;             // D^-0.5
        if (msk) v = -3.0e38f;
        v += ebias[((size_t)b * SS + q) * SS + key];
        sc[tk][r] = v;
      }
    }
    // online softmax update (row reductions over 16-lane halves)
    float al[8];
#pragma unroll
    for (int r = 0; r < 8; r++) {
      float v = fmaxf(sc[0][r], sc[1][r]);
#pragma unroll
      for (int m2 = 1; m2 < 16; m2 <<= 1) v = fmaxf(v, __shfl_xor(v, m2, 32));
      const float mn = fmaxf(ml[r], v);
      al[r] = __expf(ml[r] - mn);
      float p0 = __expf(sc[0][r] - mn);
      float p1 = __expf(sc[1][r] - mn);
      sc[0][r] = p0; sc[1][r] = p1;
      float s = p0 + p1;
#pragma unroll
      for (int m2 = 1; m2 < 16; m2 <<= 1) s += __shfl_xor(s, m2, 32);
      ll[r] = ll[r] * al[r] + s;
      ml[r] = mn;
    }
#pragma unroll
    for (int dt = 0; dt < 4; dt++)
#pragma unroll
      for (int r = 0; r < 8; r++) O[dt][r] *= al[r];

    // write P (C-layout) to per-wave LDS, reload in A-frag layout.
    // Intra-wave LDS store->load is in-order (DScnt), no WG barrier needed.
#pragma unroll
    for (int tk = 0; tk < 2; tk++)
#pragma unroll
      for (int r = 0; r < 8; r++)
        Pb[w][r + 8 * hl][tk * 16 + l15] = (bf16)sc[tk][r];
    AF pf;
    {
      const bf16* pp = &Pb[w][l15][hl * 8];
      pf.q[0] = *(const uint4*)pp;
      pf.q[1] = *(const uint4*)(pp + 16);
    }
    // O += P (16x32) @ V (32x64)
#pragma unroll
    for (int dt = 0; dt < 4; dt++) {
      AF vf;
      const bf16* vp = &Vt[dt * 16 + l15][hl * 8];
      vf.q[0] = *(const uint4*)vp;
      vf.q[1] = *(const uint4*)(vp + 16);
      O[dt] = wmma_bf16(pf.v, vf.v, O[dt]);
    }
    __syncthreads();
  }

  // normalize and store bf16 attention output [B,S,E] at (h*D + d)
#pragma unroll
  for (int r = 0; r < 8; r++) {
    const float inv = 1.0f / fmaxf(ll[r], 1e-20f);
    const int q = q0 + r + 8 * hl;
    bf16* op = outB + ((size_t)(b * SS + q)) * EE + h * DD;
#pragma unroll
    for (int dt = 0; dt < 4; dt++)
      op[dt * 16 + l15] = (bf16)(O[dt][r] * inv);
  }
}

// ---------------------------------------------------------------- host
extern "C" void kernel_launch(void* const* d_in, const int* in_sizes, int n_in,
                              void* d_out, int out_size, void* d_ws, size_t ws_size,
                              hipStream_t stream) {
  const float* src   = (const float*)d_in[0];
  const float* ebias = (const float*)d_in[1];
  const unsigned char* kpm = (const unsigned char*)d_in[2];
  const float* Wq = (const float*)d_in[3];
  const float* bq = (const float*)d_in[4];
  const float* Wk = (const float*)d_in[5];
  const float* bk = (const float*)d_in[6];
  const float* Wv = (const float*)d_in[7];
  const float* bv = (const float*)d_in[8];
  const float* Wo = (const float*)d_in[9];
  const float* bo = (const float*)d_in[10];
  const float* ln1g = (const float*)d_in[11];
  const float* ln1b = (const float*)d_in[12];
  const float* ln2g = (const float*)d_in[13];
  const float* ln2b = (const float*)d_in[14];
  const float* W1 = (const float*)d_in[15];
  const float* b1 = (const float*)d_in[16];
  const float* W2 = (const float*)d_in[17];
  const float* b2 = (const float*)d_in[18];
  float* x = (float*)d_out;  // residual stream lives in d_out

  char* p = (char*)d_ws;
  auto carve = [&](size_t elems) -> bf16* {
    bf16* r = (bf16*)p;
    p += ((elems * sizeof(bf16)) + 255) & ~(size_t)255;
    return r;
  };
  const size_t WEE = (size_t)EE * EE, WEF = (size_t)EE * FFN;
  bf16* wqB = carve((size_t)LAYERS * WEE);
  bf16* wkB = carve((size_t)LAYERS * WEE);
  bf16* wvB = carve((size_t)LAYERS * WEE);
  bf16* woB = carve((size_t)LAYERS * WEE);
  bf16* w1B = carve((size_t)LAYERS * WEF);
  bf16* w2B = carve((size_t)LAYERS * WEF);
  bf16* hB  = carve((size_t)MROWS * EE);
  bf16* qB  = carve((size_t)MROWS * EE);
  bf16* kB  = carve((size_t)MROWS * EE);
  bf16* vB  = carve((size_t)MROWS * EE);
  bf16* aB  = carve((size_t)MROWS * EE);
  bf16* fB  = carve((size_t)MROWS * FFN);

  cvt_kernel<<<512, 256, 0, stream>>>(Wq, wqB, (size_t)LAYERS * WEE);
  cvt_kernel<<<512, 256, 0, stream>>>(Wk, wkB, (size_t)LAYERS * WEE);
  cvt_kernel<<<512, 256, 0, stream>>>(Wv, wvB, (size_t)LAYERS * WEE);
  cvt_kernel<<<512, 256, 0, stream>>>(Wo, woB, (size_t)LAYERS * WEE);
  cvt_kernel<<<512, 256, 0, stream>>>(W1, w1B, (size_t)LAYERS * WEF);
  cvt_kernel<<<512, 256, 0, stream>>>(W2, w2B, (size_t)LAYERS * WEF);
  (void)hipMemcpyAsync(x, src, (size_t)MROWS * EE * sizeof(float),
                       hipMemcpyDeviceToDevice, stream);

  const dim3 gE(EE / 128, MROWS / 128);
  const dim3 gF(FFN / 128, MROWS / 128);
  for (int l = 0; l < LAYERS; l++) {
    layernorm_kernel<<<MROWS, 256, 0, stream>>>(x, ln1g + l * EE, ln1b + l * EE, hB);
    gemm_bf16_kernel<<<gE, 256, 0, stream>>>(hB, wqB + l * WEE, bq + l * EE,
                                             nullptr, nullptr, qB, MROWS, EE, EE, 0);
    gemm_bf16_kernel<<<gE, 256, 0, stream>>>(hB, wkB + l * WEE, bk + l * EE,
                                             nullptr, nullptr, kB, MROWS, EE, EE, 0);
    gemm_bf16_kernel<<<gE, 256, 0, stream>>>(hB, wvB + l * WEE, bv + l * EE,
                                             nullptr, nullptr, vB, MROWS, EE, EE, 0);
    attention_kernel<<<BB * HH * (SS / 64), 128, 0, stream>>>(qB, kB, vB, ebias, kpm, aB);
    gemm_bf16_kernel<<<gE, 256, 0, stream>>>(aB, woB + l * WEE, bo + l * EE,
                                             x, x, nullptr, MROWS, EE, EE, 0);
    layernorm_kernel<<<MROWS, 256, 0, stream>>>(x, ln2g + l * EE, ln2b + l * EE, hB);
    gemm_bf16_kernel<<<gF, 256, 0, stream>>>(hB, w1B + l * WEF, b1 + l * FFN,
                                             nullptr, nullptr, fB, MROWS, FFN, EE, 1);
    gemm_bf16_kernel<<<gE, 256, 0, stream>>>(fB, w2B + l * WEF, b2 + l * EE,
                                             x, x, nullptr, MROWS, EE, FFN, 0);
  }
}
